// S4Layer_16741782520105
// MI455X (gfx1250) — matrix operand order repaired and checked by reading, not compiled, not run on hardware
//
#include <hip/hip_runtime.h>

// ---------------------------------------------------------------------------
// S4 layer via chunked SSM scan (rank-64 kernel), CDNA5 / gfx1250, wave32.
//
//   y[b,l,d] = sum_{k<=l} K[d,k] u[b,l-k,d] + D[d] u[b,l,d]
//   K[d,l]   = sum_n coeff[d,n] * ab[d,n]^l          (bilinear discretization)
//
// Chunked form (T = 64, 64 chunks):
//   y_chunk = Tri(64x64) @ u_chunk  +  P(64x64) @ S        (two WMMA matmuls)
//   S_new   = a^T (.) S  +  Qt(64x64) @ u_chunk            (one WMMA matmul)
//
// One block per channel d (512 blocks x 128 threads = 4 waves). N dim of the
// WMMA = batch (8 real cols, cols 8..15 zero). bf16 operands / f32 accum via
// v_wmma_f32_16x16x32_bf16. ~3.4 G MACs total -> memory bound; the whole
// working set (~128MB) sits in the 192MB L2.
//
// v2: double-buffered ASYNC global->LDS DMA (global_load_async_to_lds_b32,
//     ASYNCcnt, s_wait_asynccnt) overlaps chunk c+1's global latency with
//     chunk c's WMMA work; D*u reads the staged f32 from LDS.
// v3: loop-invariant A-fragments (Tri/P/Qt) hoisted into registers (removes
//     12 ds_load_b128 per chunk/wave); f32 master state kept in registers
//     (each thread owns its accS fragment rows) -> state update is 8 FMAs +
//     one 16B ds_store_b128, Sf/Supd LDS arrays deleted.
// ---------------------------------------------------------------------------

typedef __attribute__((ext_vector_type(16))) __bf16 v16bf;
typedef __attribute__((ext_vector_type(8)))  __bf16 v8bf;
typedef __attribute__((ext_vector_type(8)))  float  v8f;

#define D_MODEL  512
#define NSTATE   64
#define LMAX     4096
#define CHUNK    64
#define NCHUNK   (LMAX / CHUNK)
#define BATCH    8
#define NTHREADS 128

// A-matrix fragment (16x32 bf16), source row-major [64x64] in LDS.
// Layout: lanes 0-15 row m0+lane, K=k0..k0+15 contiguous;
// lanes 16-31 K=k0+16..k0+31 (matches load_b_frag_T).
__device__ __forceinline__ v16bf load_a_frag(const __bf16* M, int m0, int k0, int lane) {
  const int row = m0 + (lane & 15);
  const int kb  = k0 + ((lane & 16) ? 16 : 0);
  v8bf lo = *(const v8bf*)(M + row * 64 + kb);        // 16B contiguous
  v8bf hi = *(const v8bf*)(M + row * 64 + kb + 8);    // 16B contiguous
  return __builtin_shufflevector(lo, hi, 0,1,2,3,4,5,6,7,8,9,10,11,12,13,14,15);
}

// B-matrix fragment (32x16 bf16), source stored TRANSPOSED [16 cols][64 rows]
// so each lane's 16 K-values are contiguous. Layout: lanes 0-15 col=lane,
// K=k0..k0+15; lanes 16-31 col=lane-16, K=k0+16..k0+31.
__device__ __forceinline__ v16bf load_b_frag_T(const __bf16* MT, int k0, int lane) {
  const int col = lane & 15;
  const int kb  = k0 + ((lane & 16) ? 16 : 0);
  return *(const v16bf*)(MT + col * 64 + kb);         // 32B contiguous
}

// Generic (flat) pointer to a __shared__ object: low 32 bits are the LDS
// offset within the workgroup allocation (flat aperture encoding).
__device__ __forceinline__ unsigned lds_off32(const void* p) {
  return (unsigned)(unsigned long long)p;
}

// Issue async DMA of one u chunk (512 scattered f32) into LDS buffer dstUf.
// GVS addressing: 64-bit SGPR base + per-lane signed 32-bit byte offset
// (max offset 7*8MB+8MB < 2^31). Tracked by ASYNCcnt.
__device__ __forceinline__ void issue_async_chunk(const float* __restrict__ u,
                                                  int d, int cc,
                                                  float* dstUf, int tid) {
  #pragma unroll
  for (int j = tid; j < CHUNK * BATCH; j += NTHREADS) {   // 4 issues/thread
    const int t = j >> 3, b = j & 7;
    const int goff = (int)(((unsigned)b * (LMAX * D_MODEL) +
                            (unsigned)(cc * CHUNK + t) * D_MODEL +
                            (unsigned)d) * sizeof(float));
    const unsigned ldsa = lds_off32(dstUf + j);
    asm volatile("global_load_async_to_lds_b32 %0, %1, %2"
                 :: "v"(ldsa), "v"(goff), "s"(u)
                 : "memory");
  }
}

__device__ __forceinline__ void wait_async_all() {
  asm volatile("s_wait_asynccnt 0x0" ::: "memory");
}

__global__ __launch_bounds__(NTHREADS)
void s4_chunked_scan_kernel(const float* __restrict__ u,
                            const float* __restrict__ Bmat,
                            const float* __restrict__ Cmat,
                            const float* __restrict__ Dvec,
                            const float* __restrict__ stepv,
                            const float* __restrict__ adiag,
                            float* __restrict__ out) {
  __shared__ __bf16 Tri[64 * 64];          // intra-chunk Toeplitz (lower-tri)
  __shared__ __bf16 Pm [64 * 64];          // inter-chunk projector  P[t][n]
  __shared__ __bf16 Qt [64 * 64];          // state gatherer        Qt[n][k]
  __shared__ __bf16 UT [16 * 64];          // u chunk bf16, transposed [col][t]
  __shared__ __bf16 ST [16 * 64];          // state bf16, transposed [col][n]
  __shared__ float  Uf [2][CHUNK * BATCH]; // u chunk f32, double-buffered [t*8+b]
  __shared__ float  Apow[(CHUNK + 1) * NSTATE];
  __shared__ float  coeff_s[NSTATE];
  __shared__ float  aT_s[NSTATE];
  __shared__ float  Kin[CHUNK];

  const int d    = blockIdx.x;
  const int tid  = threadIdx.x;
  const int lane = tid & 31;
  const int m0    = (tid >> 5) * 16;            // wave's M-tile origin
  const int rbase = m0 + ((lane & 16) ? 8 : 0); // this lane's first C/D row
  const int colc  = lane & 15;                  // this lane's C/D column

  // Kick off the DMA for chunk 0 first; matrix setup below overlaps it.
  issue_async_chunk(u, d, 0, Uf[0], tid);

  // ---- per-channel discretization + power table (recomputed per block) ----
  const float stepd = stepv[d];
  if (tid < NSTATE) {
    const int n   = tid;
    const float a = adiag[n];
    const float h = 0.5f * stepd * a;
    const float bl = 1.0f / (1.0f - h);
    const float ab = bl * (1.0f + h);
    const float bb = stepd * bl * Bmat[n * D_MODEL + d];
    coeff_s[n] = Cmat[d * NSTATE + n] * bb;
    float p = 1.0f;
    for (int t = 0; t <= CHUNK; ++t) { Apow[t * NSTATE + n] = p; p *= ab; }
    aT_s[n] = Apow[CHUNK * NSTATE + n];
  }
  __syncthreads();

  if (tid < CHUNK) {                       // Kin[tau] = sum_n coeff * ab^tau
    float s = 0.f;
    #pragma unroll 8
    for (int n = 0; n < NSTATE; ++n) s += coeff_s[n] * Apow[tid * NSTATE + n];
    Kin[tid] = s;
  }
  __syncthreads();

  for (int i = tid; i < 64 * 64; i += NTHREADS) {
    const int r = i >> 6, c = i & 63;
    Tri[i] = (r >= c) ? (__bf16)Kin[r - c] : (__bf16)0.f;
    Pm[i]  = (__bf16)(coeff_s[c] * Apow[(r + 1) * NSTATE + c]);
    Qt[i]  = (__bf16)Apow[(CHUNK - 1 - c) * NSTATE + r];
  }
  for (int i = tid; i < 16 * 64; i += NTHREADS) {
    ST[i] = (__bf16)0.f; UT[i] = (__bf16)0.f;   // pad cols stay 0 forever
  }
  __syncthreads();

  // ---- hoist loop-invariant operands into registers --------------------
  const v16bf aT0 = load_a_frag(Tri, m0, 0,  lane);
  const v16bf aT1 = load_a_frag(Tri, m0, 32, lane);
  const v16bf aP0 = load_a_frag(Pm,  m0, 0,  lane);
  const v16bf aP1 = load_a_frag(Pm,  m0, 32, lane);
  const v16bf aQ0 = load_a_frag(Qt,  m0, 0,  lane);
  const v16bf aQ1 = load_a_frag(Qt,  m0, 32, lane);

  float aT_r[8];                 // per-row chunk-decay factors a_n^T
  float S_reg[8];                // f32 master state, rows rbase..rbase+7, col colc
  #pragma unroll
  for (int r = 0; r < 8; ++r) { aT_r[r] = aT_s[rbase + r]; S_reg[r] = 0.f; }

  const float  Dd      = Dvec[d];
  const size_t strideB = (size_t)LMAX * D_MODEL;

  // -------------------------- sequential chunk scan -----------------------
  for (int c = 0; c < NCHUNK; ++c) {
    const int buf = c & 1;

    wait_async_all();        // my chunk-c DMA done (async loads retire in order)
    __syncthreads();         // everyone's DMA done; iteration c-1 fully retired

    // Safe now to overwrite buf^1: issue DMA for chunk c+1 (overlaps compute).
    if (c + 1 < NCHUNK)
      issue_async_chunk(u, d, c + 1, Uf[buf ^ 1], tid);

    // convert staged f32 chunk -> bf16 transposed WMMA-B operand
    for (int j = tid; j < CHUNK * BATCH; j += NTHREADS) {
      const int t = j >> 3, b = j & 7;
      UT[b * 64 + t] = (__bf16)Uf[buf][j];
    }
    __syncthreads();

    v8f accY = {};
    v8f accS = {};
    {
      v16bf bU0 = load_b_frag_T(UT, 0,  lane);
      v16bf bS0 = load_b_frag_T(ST, 0,  lane);
      v16bf bU1 = load_b_frag_T(UT, 32, lane);
      v16bf bS1 = load_b_frag_T(ST, 32, lane);
      accY = __builtin_amdgcn_wmma_f32_16x16x32_bf16(false, aT0, false, bU0, (short)0, accY, false, false);
      accY = __builtin_amdgcn_wmma_f32_16x16x32_bf16(false, aP0, false, bS0, (short)0, accY, false, false);
      accS = __builtin_amdgcn_wmma_f32_16x16x32_bf16(false, aQ0, false, bU0, (short)0, accS, false, false);
      accY = __builtin_amdgcn_wmma_f32_16x16x32_bf16(false, aT1, false, bU1, (short)0, accY, false, false);
      accY = __builtin_amdgcn_wmma_f32_16x16x32_bf16(false, aP1, false, bS1, (short)0, accY, false, false);
      accS = __builtin_amdgcn_wmma_f32_16x16x32_bf16(false, aQ1, false, bU1, (short)0, accS, false, false);
    }
    __syncthreads();   // all waves finished reading ST before it is refreshed

    // In-register state update: S = a^T (.) S + Qt@u.  Each thread owns the
    // 8 state elements matching its accS fragment (rows rbase..rbase+7, col
    // colc), so only the bf16 operand copy touches LDS: 8 contiguous bf16
    // = one 16B ds_store_b128.
    v8bf stv;
    #pragma unroll
    for (int r = 0; r < 8; ++r) {
      S_reg[r] = aT_r[r] * S_reg[r] + accS[r];
      stv[r]   = (__bf16)S_reg[r];
    }
    *(v8bf*)(ST + colc * 64 + rbase) = stv;

    // y = accY + D*u; D*u term reads the staged f32 chunk from LDS (buf is
    // untouched until iteration c+1's DMA targets it again).
    #pragma unroll
    for (int r = 0; r < 8; ++r) {
      const int t = rbase + r;
      if (colc < BATCH) {
        const size_t gidx = (size_t)colc * strideB + (size_t)(c * CHUNK + t) * D_MODEL + d;
        out[gidx] = accY[r] + Dd * Uf[buf][t * 8 + colc];
      }
    }
    // next iteration's top wait+barrier orders UT/ST/Uf reuse
  }
}

extern "C" void kernel_launch(void* const* d_in, const int* in_sizes, int n_in,
                              void* d_out, int out_size, void* d_ws, size_t ws_size,
                              hipStream_t stream) {
  (void)in_sizes; (void)n_in; (void)d_ws; (void)ws_size; (void)out_size;
  const float* u     = (const float*)d_in[0];  // [8, 4096, 512]
  const float* Bmat  = (const float*)d_in[1];  // [64, 512]
  const float* Cmat  = (const float*)d_in[2];  // [512, 64]
  const float* Dvec  = (const float*)d_in[3];  // [1,1,512] -> 512
  const float* stepv = (const float*)d_in[4];  // [512]
  const float* adiag = (const float*)d_in[5];  // [64]
  float* out = (float*)d_out;                  // [8, 4096, 512]

  s4_chunked_scan_kernel<<<dim3(D_MODEL), dim3(NTHREADS), 0, stream>>>(
      u, Bmat, Cmat, Dvec, stepv, adiag, out);
}
